// RelationModel_32538672235024
// MI455X (gfx1250) — matrix operand order repaired and checked by reading, not compiled, not run on hardware
//
#include <hip/hip_runtime.h>
#include <math.h>

// ---- problem constants (match reference) ----
constexpr int kS     = 512;
constexpr int kH     = 1024;
constexpr int kP     = 32768;
constexpr int kR     = 12;
constexpr int kKTOT  = 3 * kH;          // 3072
constexpr int kNOUT  = kH / 2;          // 512
constexpr int kNTILES = kNOUT / 16;     // 32 n-tiles of 16

typedef __attribute__((ext_vector_type(16))) __bf16 v16bf;
typedef __attribute__((ext_vector_type(8)))  float  v8f;

__device__ __forceinline__ unsigned int pack2_bf16(float a, float b) {
  unsigned int ua = __float_as_uint(a), ub = __float_as_uint(b);
  ua = (ua + 0x7FFFu + ((ua >> 16) & 1u)) >> 16;      // RNE fp32 -> bf16
  ub = (ub + 0x7FFFu + ((ub >> 16) & 1u)) >> 16;
  return (ua & 0xFFFFu) | (ub << 16);
}

__device__ __forceinline__ unsigned short f32_to_bf16(float f) {
  unsigned int u = __float_as_uint(f);
  return (unsigned short)((u + 0x7FFFu + ((u >> 16) & 1u)) >> 16);
}

// ---------------------------------------------------------------------------
// Kernel 1: W1 fp32 [3072 x 512] -> bf16, pre-swizzled into WMMA B-register
// layout. Blocks of 1 KB per (ktile, ntile); within a block: lane*32B, then
// VGPR v (4B), then low/high 16-bit half.
// B layout: lanes 0-15: K=0..7 (v0-3), 16..23 (v4-7), N=lane; lanes 16-31: K=8..15, 24..31.
// ---------------------------------------------------------------------------
__global__ __launch_bounds__(256) void w1_swizzle_bf16(
    const float* __restrict__ W1, unsigned short* __restrict__ W1b) {
  int idx = blockIdx.x * 256 + threadIdx.x;   // 3072*512 total
  int k = idx >> 9;          // row of W1 (K index)
  int n = idx & 511;         // col of W1 (N index)
  unsigned short hv = f32_to_bf16(W1[idx]);
  int ktile = k >> 5, kin = k & 31;
  int ntile = n >> 4, nl  = n & 15;
  int lane = ((kin & 8) << 1) + nl;      // 0-15 or 16-31
  int v    = ((kin & 7) + ((kin & 16) >> 1)) >> 1;   // VGPR index 0..7
  int hi   = kin & 1;                    // low/high 16 bits
  size_t off = (((size_t)ktile * kNTILES + ntile) << 10)
             + (size_t)lane * 32 + (size_t)v * 4 + (size_t)hi * 2;
  *(unsigned short*)((char*)W1b + off) = hv;
}

// ---------------------------------------------------------------------------
// Kernel 2: fused gather + GEMM1 (bf16 WMMA, fp32 acc) + bias + exact GELU +
// GEMM2 (fp32) + softmax.
// Block = 32 pairs, 8 waves. Wave (mg, ng): m-group mg = wave>>2 (16 rows),
// n-group ng = wave&3 (8 n-tiles of 16 -> N=128). 8 WMMA per K-step per wave.
// ---------------------------------------------------------------------------
__global__ __launch_bounds__(256) void relation_fused(
    const float* __restrict__ hs,  const int* __restrict__ pair_b,
    const int* __restrict__ idx1,  const int* __restrict__ idx2,
    const unsigned short* __restrict__ W1b, const float* __restrict__ b1,
    const float* __restrict__ W2,  const float* __restrict__ b2,
    float* __restrict__ out) {

  __shared__ unsigned short h16[32][kNOUT + 8];   // bf16 h, padded stride
  __shared__ float part[32][16][kR];              // GEMM2 partial sums
  __shared__ float lg[32][kR];
  __shared__ float smax[32], ssum[32];

  const int tid  = threadIdx.x;
  const int wave = tid >> 5;
  const int lane = tid & 31;
  const int mg   = wave >> 2;      // m-group: 0 or 1
  const int ng   = wave & 3;       // n-group: n-tiles [ng*8, ng*8+8)
  const int m    = lane & 15;      // row within the 16-row m-group
  const int half = lane >> 4;      // A-layout half (K grouping)
  const int tile = blockIdx.x;     // 1024 tiles of 32 pairs
  const int p    = tile * 32 + mg * 16 + m;

  // Per-lane gather bases for this pair's three source rows.
  const int b  = pair_b[p];
  const int i1 = idx1[p];
  const int i2 = idx2[p];
  const float* base0 = hs + ((size_t)b * kS) * kH;           // CLS token
  const float* base1 = hs + ((size_t)b * kS + i1) * kH;      // entity 1
  const float* base2 = hs + ((size_t)b * kS + i2) * kH;      // entity 2

  v8f acc[8];
#pragma unroll
  for (int t = 0; t < 8; ++t) acc[t] = v8f{};

  const int nblk0 = ng * 8;        // first of this wave's 8 n-tiles

#pragma unroll
  for (int seg = 0; seg < 3; ++seg) {            // cls / e1 / e2 segments
    const float* bp = (seg == 0 ? base0 : (seg == 1 ? base1 : base2)) + half * 8;
    const char* wbseg = (const char*)W1b
                      + (((size_t)(seg * 32) * kNTILES + nblk0) << 10)
                      + (size_t)lane * 32;
    for (int cc = 0; cc < 32; ++cc) {            // 32 K-chunks per segment
      // ---- A raw loads (gather): lanes 0-15 K{0..7,16..23}, 16-31 K{8..15,24..31}
      const float* rp = bp + (cc << 5);
      float4 f0 = *(const float4*)(rp);
      float4 f1 = *(const float4*)(rp + 4);
      float4 f2 = *(const float4*)(rp + 16);
      float4 f3 = *(const float4*)(rp + 20);

      // ---- B loads: all 16 x b128 issued together (one clause) ----
      const char* wb = wbseg + (((size_t)cc * kNTILES) << 10);
      int4 q[16];
#pragma unroll
      for (int t = 0; t < 8; ++t) {
        q[2 * t]     = *(const int4*)(wb + (size_t)t * 1024);
        q[2 * t + 1] = *(const int4*)(wb + (size_t)t * 1024 + 16);
      }
      __builtin_prefetch(wb + ((size_t)kNTILES << 10), 0, 0);  // next K-chunk

      // ---- pack A fp32 -> bf16 (overlaps in-flight B loads) ----
      union { unsigned int u[8]; v16bf v; } A;
      A.u[0] = pack2_bf16(f0.x, f0.y); A.u[1] = pack2_bf16(f0.z, f0.w);
      A.u[2] = pack2_bf16(f1.x, f1.y); A.u[3] = pack2_bf16(f1.z, f1.w);
      A.u[4] = pack2_bf16(f2.x, f2.y); A.u[5] = pack2_bf16(f2.z, f2.w);
      A.u[6] = pack2_bf16(f3.x, f3.y); A.u[7] = pack2_bf16(f3.z, f3.w);

      // ---- 8 WMMAs reusing one A fragment ----
#pragma unroll
      for (int t = 0; t < 8; ++t) {
        union { int4 q2[2]; v16bf v; } B;
        B.q2[0] = q[2 * t]; B.q2[1] = q[2 * t + 1];
        acc[t] = __builtin_amdgcn_wmma_f32_16x16x32_bf16(
            false, A.v, false, B.v, (short)0, acc[t], false, false);
      }
    }
  }

  // ---- bias + exact GELU, deposit h (bf16) into LDS ----
  // C/D layout: VGPR j holds M=j (lanes 0-15) / M=8+j (lanes 16-31), N=lane%16.
#pragma unroll
  for (int t = 0; t < 8; ++t) {
    int n = (nblk0 + t) * 16 + m;
    float bias = b1[n];
#pragma unroll
    for (int j = 0; j < 8; ++j) {
      int mm = mg * 16 + j + half * 8;
      float x = acc[t][j] + bias;
      float g = 0.5f * x * (1.0f + erff(x * 0.70710678118654752f));
      h16[mm][n] = f32_to_bf16(g);
    }
  }
  __syncthreads();

  // ---- GEMM2 partials: 512 (row, n-chunk) tasks, 32-length dots ----
  {
    const int nc = tid & 15;                 // n-chunk of 32
#pragma unroll
    for (int rep = 0; rep < 2; ++rep) {
      const int mm = (tid >> 4) + rep * 16;
      float a[kR];
#pragma unroll
      for (int r = 0; r < kR; ++r) a[r] = 0.f;
      const unsigned short* hrow = &h16[mm][nc * 32];
      const float* w2p = W2 + (size_t)nc * 32 * kR;
      for (int i4 = 0; i4 < 8; ++i4) {       // 4 n's per step
        uint2 hq = *(const uint2*)(hrow + i4 * 4);
        float hv0 = __uint_as_float(hq.x << 16);
        float hv1 = __uint_as_float(hq.x & 0xFFFF0000u);
        float hv2 = __uint_as_float(hq.y << 16);
        float hv3 = __uint_as_float(hq.y & 0xFFFF0000u);
        union W12 { float4 q[3]; float f[12]; } w0, w1, w2v, w3;
        const float* wr = w2p + (size_t)i4 * 4 * kR;
        w0.q[0] = *(const float4*)(wr);          w0.q[1] = *(const float4*)(wr + 4);
        w0.q[2] = *(const float4*)(wr + 8);
        w1.q[0] = *(const float4*)(wr + 12);     w1.q[1] = *(const float4*)(wr + 16);
        w1.q[2] = *(const float4*)(wr + 20);
        w2v.q[0] = *(const float4*)(wr + 24);    w2v.q[1] = *(const float4*)(wr + 28);
        w2v.q[2] = *(const float4*)(wr + 32);
        w3.q[0] = *(const float4*)(wr + 36);     w3.q[1] = *(const float4*)(wr + 40);
        w3.q[2] = *(const float4*)(wr + 44);
#pragma unroll
        for (int r = 0; r < kR; ++r)
          a[r] += hv0 * w0.f[r] + hv1 * w1.f[r] + hv2 * w2v.f[r] + hv3 * w3.f[r];
      }
#pragma unroll
      for (int r = 0; r < kR; ++r) part[mm][nc][r] = a[r];
    }
  }
  __syncthreads();

  // ---- reduce partials -> logits ----
  for (int t = tid; t < 32 * kR; t += 256) {
    int mm = t & 31, r = t >> 5;
    float s = b2[r];
#pragma unroll
    for (int ncc = 0; ncc < 16; ++ncc) s += part[mm][ncc][r];
    lg[mm][r] = s;
  }
  __syncthreads();

  // ---- softmax statistics ----
  if (tid < 32) {
    float mx = lg[tid][0];
#pragma unroll
    for (int r = 1; r < kR; ++r) mx = fmaxf(mx, lg[tid][r]);
    float s = 0.f;
#pragma unroll
    for (int r = 0; r < kR; ++r) s += __expf(lg[tid][r] - mx);
    smax[tid] = mx; ssum[tid] = s;
  }
  __syncthreads();

  // ---- write logits then probs (tuple concatenated flat) ----
  for (int t = tid; t < 32 * kR; t += 256) {
    int mm = t & 31, r = t >> 5;
    size_t pglob = (size_t)tile * 32 + mm;
    float l = lg[mm][r];
    out[pglob * kR + r] = l;
    out[(size_t)kP * kR + pglob * kR + r] = __expf(l - smax[mm]) / ssum[mm];
  }
}

// ---------------------------------------------------------------------------
extern "C" void kernel_launch(void* const* d_in, const int* in_sizes, int n_in,
                              void* d_out, int out_size, void* d_ws, size_t ws_size,
                              hipStream_t stream) {
  const float* hs     = (const float*)d_in[0];
  const int*   pair_b = (const int*)d_in[1];
  const int*   i1     = (const int*)d_in[2];
  const int*   i2     = (const int*)d_in[3];
  const float* W1     = (const float*)d_in[4];
  const float* b1     = (const float*)d_in[5];
  const float* W2     = (const float*)d_in[6];
  const float* b2     = (const float*)d_in[7];

  unsigned short* W1b = (unsigned short*)d_ws;   // 3072*512*2 = 3 MB scratch

  hipLaunchKernelGGL(w1_swizzle_bf16, dim3((kKTOT * kNOUT) / 256), dim3(256),
                     0, stream, W1, W1b);
  hipLaunchKernelGGL(relation_fused, dim3(kP / 32), dim3(256), 0, stream,
                     hs, pair_b, i1, i2, W1b, b1, W2, b2, (float*)d_out);
}
